// MultiHeadSelfAttention_11450382811213
// MI455X (gfx1250) — matrix operand order repaired and checked by reading, not compiled, not run on hardware
//
#include <hip/hip_runtime.h>

#define BATCH  2
#define SEQ    2048
#define DMODEL 1024
#define NHEAD  16
#define DHEAD  64

typedef __bf16 bf16_t;
typedef bf16_t          v16bf __attribute__((ext_vector_type(16)));
typedef float           v8f   __attribute__((ext_vector_type(8)));
typedef unsigned short  v16u  __attribute__((ext_vector_type(16)));
typedef unsigned int    v4u   __attribute__((ext_vector_type(4)));
typedef int             v8i   __attribute__((ext_vector_type(8)));
typedef int             v4i   __attribute__((ext_vector_type(4)));

// Tensor Data Mover availability (this toolchain: 6-arg builtin, clang-23 style).
#if defined(__HIP_DEVICE_COMPILE__) && defined(__has_builtin)
#  if __has_builtin(__builtin_amdgcn_tensor_load_to_lds) && \
      __has_builtin(__builtin_amdgcn_s_wait_tensorcnt)
#    define USE_TDM 1
#  endif
#endif
#ifndef USE_TDM
#  define USE_TDM 0
#endif

// ---------- helpers ----------

__device__ __forceinline__ unsigned short f32_to_bf16_rne(float f) {
    unsigned int u = __builtin_bit_cast(unsigned int, f);
    u += 0x7FFFu + ((u >> 16) & 1u);           // round-to-nearest-even
    return (unsigned short)(u >> 16);
}

// A-fragment (16x32 bf16, MxK).  lane(0..15)=row M, g=lane>>4,
// VGPR v holds K = g*16 + v*2 (+1 in high half).
__device__ __forceinline__ v16bf load_a16x32(const unsigned short* __restrict__ p,
                                             int ld, int m0, int k0, int lane) {
    const int m = m0 + (lane & 15);
    const int g = (lane >> 4) & 1;
    const unsigned short* base = p + (size_t)m * ld + k0 + g * 16;
    v16u r;
#pragma unroll
    for (int v = 0; v < 8; ++v) {
        const int k = v << 1;
        const unsigned int pair = *(const unsigned int*)(base + k);
        r[2 * v]     = (unsigned short)(pair & 0xFFFFu);
        r[2 * v + 1] = (unsigned short)(pair >> 16);
    }
    return __builtin_bit_cast(v16bf, r);
}

// B-fragment (32x16 bf16, KxN) from memory stored [n][k]: lane=col N, halves e hold K=g*16+e.
__device__ __forceinline__ v16bf load_b_nk(const unsigned short* __restrict__ p,
                                           int ld, int n0, int k0, int lane) {
    const int n = n0 + (lane & 15);
    const int g = (lane >> 4) & 1;
    const unsigned short* base = p + (size_t)n * ld + k0 + g * 16;
    v16u r;
#pragma unroll
    for (int v = 0; v < 8; ++v) {
        const unsigned int pair = *(const unsigned int*)(base + 2 * v);
        r[2 * v]     = (unsigned short)(pair & 0xFFFFu);
        r[2 * v + 1] = (unsigned short)(pair >> 16);
    }
    return __builtin_bit_cast(v16bf, r);
}

// B-fragment (32x16) from memory stored [k][n] row-major (V: [key][dh]).
__device__ __forceinline__ v16bf load_b_kn(const unsigned short* __restrict__ p,
                                           int ld, int k0, int n0, int lane) {
    const int n = n0 + (lane & 15);
    const int g = (lane >> 4) & 1;
    v16u r;
#pragma unroll
    for (int e = 0; e < 16; ++e)
        r[e] = p[(size_t)(k0 + g * 16 + e) * ld + n];
    return __builtin_bit_cast(v16bf, r);
}

__device__ __forceinline__ v8f wmma_bf16(v16bf a, v16bf b, v8f c) {
    return __builtin_amdgcn_wmma_f32_16x16x32_bf16(false, a, false, b,
                                                   (short)0, c, false, false);
}

#if USE_TDM
// Issue a TDM 2D tile load: tensor is (tensor_rows x K) bf16 row-major,
// tile is (rows x cols) at (row0, k0), destination LDS offset lds_off.
// D# packing per CDNA5 ISA.
__device__ __forceinline__ void tdm_load_2d(const unsigned short* gbase,
                                            int row0, int k0, int K_,
                                            int rows, int cols, int tensor_rows,
                                            unsigned lds_off) {
    const unsigned long long ga =
        (unsigned long long)(const void*)(gbase + (size_t)row0 * K_ + k0);
    v4u g0;
    g0[0] = 1u;                                                 // count=1, user mode
    g0[1] = lds_off;                                            // lds_addr (bytes)
    g0[2] = (unsigned)(ga & 0xFFFFFFFFull);                     // global_addr[31:0]
    g0[3] = (unsigned)((ga >> 32) & 0x1FFFFFFull) | (2u << 30); // addr[56:32] | type=2
    const unsigned long long s0 = (unsigned long long)K_;       // dim0 stride (elements)
    v8i g1;
    g1[0] = (int)(1u << 16);                                    // data_size=1 (2 bytes)
    g1[1] = (int)(((unsigned)K_ & 0xFFFFu) << 16);              // tensor_dim0[15:0]
    g1[2] = (int)(((unsigned)K_ >> 16) |
                  (((unsigned)tensor_rows & 0xFFFFu) << 16));   // dim0 hi | dim1 lo
    g1[3] = (int)(((unsigned)tensor_rows >> 16) |
                  ((unsigned)cols << 16));                      // dim1 hi | tile_dim0
    g1[4] = (int)(unsigned)rows;                                // tile_dim1 (tile_dim2=0)
    g1[5] = (int)(unsigned)(s0 & 0xFFFFFFFFull);                // dim0_stride[31:0]
    g1[6] = (int)(unsigned)((s0 >> 32) & 0xFFFFull);            // dim0_stride[47:32]
    g1[7] = 0;
    const v4i gz4 = {0, 0, 0, 0};
    const v8i gz8 = {0, 0, 0, 0, 0, 0, 0, 0};
    __builtin_amdgcn_tensor_load_to_lds(g0, g1, gz4, gz4, gz8, 0);
}
#endif

// ---------- kernels ----------

__global__ void __launch_bounds__(256) cvt_bf16_kernel(const float* __restrict__ in,
                                                       unsigned short* __restrict__ out,
                                                       int n) {
    const int i = blockIdx.x * blockDim.x + threadIdx.x;
    if (i < n) out[i] = f32_to_bf16_rne(in[i]);
}

// Y[m][n] = scale * sum_k X[m][k] * W[n][k]   (X: MxK, W: NxK bf16 row-major).
// Block = 4 waves, 128x64 output tile.  X/W K-slabs double-buffered in LDS,
// filled by the Tensor Data Mover (wave 0) and consumed by all waves.
__global__ void __launch_bounds__(128) gemm_bt_kernel(
        const unsigned short* __restrict__ X,
        const unsigned short* __restrict__ W,
        unsigned short* __restrict__ Ybf,   // bf16 output (or null)
        float* __restrict__ Yf,             // f32 output (or null)
        int M, int N, int K, float scale) {
    __shared__ unsigned short ldsX[2][128 * 32];   // 16 KiB
    __shared__ unsigned short ldsW[2][64 * 32];    //  8 KiB
    const int tid  = threadIdx.x;
    const int lane = tid & 31;
    const int wave = tid >> 5;
    const int nblk = blockIdx.x * 64;
    const int mblk = blockIdx.y * 128;
    const int wm = (wave >> 1) * 64;   // wave's rows within block tile
    const int wn = (wave & 1) * 32;    // wave's cols within block tile

    v8f acc[4][2];
#pragma unroll
    for (int i = 0; i < 4; ++i)
#pragma unroll
        for (int j = 0; j < 2; ++j) acc[i][j] = (v8f){};

    const int ksteps = K / 32;
    int cur = 0;

#if USE_TDM
    if (wave == 0) {
        tdm_load_2d(X, mblk, 0, K, 128, 32, M, (unsigned)(size_t)&ldsX[0][0]);
        tdm_load_2d(W, nblk, 0, K,  64, 32, N, (unsigned)(size_t)&ldsW[0][0]);
    }
#else
    {   // cooperative copy, 16B per thread per step
        const int k0 = 0;
#pragma unroll
        for (int t = 0; t < 4; ++t) {
            const int idx = tid + t * 128;          // 512 x 16B for X
            const int row = idx >> 2, cg = (idx & 3) * 8;
            *(uint4*)&ldsX[0][row * 32 + cg] =
                *(const uint4*)&X[(size_t)(mblk + row) * K + k0 + cg];
        }
#pragma unroll
        for (int t = 0; t < 2; ++t) {
            const int idx = tid + t * 128;          // 256 x 16B for W
            const int row = idx >> 2, cg = (idx & 3) * 8;
            *(uint4*)&ldsW[0][row * 32 + cg] =
                *(const uint4*)&W[(size_t)(nblk + row) * K + k0 + cg];
        }
    }
#endif

    for (int ks = 0; ks < ksteps; ++ks) {
#if USE_TDM
        if (wave == 0) __builtin_amdgcn_s_wait_tensorcnt(0);
#endif
        __syncthreads();   // current buffer ready for all waves
        if (ks + 1 < ksteps) {
            const int k0 = (ks + 1) * 32;
            const int nxt = cur ^ 1;
#if USE_TDM
            if (wave == 0) {
                tdm_load_2d(X, mblk, k0, K, 128, 32, M, (unsigned)(size_t)&ldsX[nxt][0]);
                tdm_load_2d(W, nblk, k0, K,  64, 32, N, (unsigned)(size_t)&ldsW[nxt][0]);
            }
#else
#pragma unroll
            for (int t = 0; t < 4; ++t) {
                const int idx = tid + t * 128;
                const int row = idx >> 2, cg = (idx & 3) * 8;
                *(uint4*)&ldsX[nxt][row * 32 + cg] =
                    *(const uint4*)&X[(size_t)(mblk + row) * K + k0 + cg];
            }
#pragma unroll
            for (int t = 0; t < 2; ++t) {
                const int idx = tid + t * 128;
                const int row = idx >> 2, cg = (idx & 3) * 8;
                *(uint4*)&ldsW[nxt][row * 32 + cg] =
                    *(const uint4*)&W[(size_t)(nblk + row) * K + k0 + cg];
            }
#endif
        }
        // compute 64x32 wave tile from LDS
        const unsigned short* xt = &ldsX[cur][0];
        const unsigned short* wt = &ldsW[cur][0];
        v16bf bfrag[2];
#pragma unroll
        for (int j = 0; j < 2; ++j)
            bfrag[j] = load_b_nk(wt, 32, wn + j * 16, 0, lane);
#pragma unroll
        for (int i = 0; i < 4; ++i) {
            const v16bf afrag = load_a16x32(xt, 32, wm + i * 16, 0, lane);
#pragma unroll
            for (int j = 0; j < 2; ++j)
                acc[i][j] = wmma_bf16(afrag, bfrag[j], acc[i][j]);
        }
        __syncthreads();   // everyone done with cur before it is refilled
        cur ^= 1;
    }

    const int g  = (lane >> 4) & 1;
    const int nl = lane & 15;
#pragma unroll
    for (int i = 0; i < 4; ++i) {
#pragma unroll
        for (int j = 0; j < 2; ++j) {
#pragma unroll
            for (int r = 0; r < 8; ++r) {
                const int m = mblk + wm + i * 16 + r + 8 * g;
                const int n = nblk + wn + j * 16 + nl;
                const float v = acc[i][j][r] * scale;
                if (Ybf) Ybf[(size_t)m * N + n] = f32_to_bf16_rne(v);
                else     Yf [(size_t)m * N + n] = v;
            }
        }
    }
}

// Causal flash attention.  Grid: (SEQ/16, BATCH*NHEAD), 1 wave / block.
// Q pre-scaled by 1/sqrt(dh).  Q,K,V,O are (B*L, D) bf16, head h at cols [h*64, h*64+64).
__global__ void __launch_bounds__(32) attn_kernel(
        const unsigned short* __restrict__ Q,
        const unsigned short* __restrict__ Kb,
        const unsigned short* __restrict__ Vb,
        unsigned short* __restrict__ O) {
    __shared__ unsigned short pbuf[16 * 32];   // P tile staging (C-layout -> A-layout)
    const int lane = threadIdx.x & 31;
    const int m0 = blockIdx.x * 16;
    const int bh = blockIdx.y;
    const int b = bh / NHEAD, h = bh % NHEAD;
    const size_t headoff = (size_t)b * SEQ * DMODEL + (size_t)h * DHEAD;
    const unsigned short* Qh = Q  + headoff;
    const unsigned short* Kh = Kb + headoff;
    const unsigned short* Vh = Vb + headoff;
    const int g  = (lane >> 4) & 1;
    const int nl = lane & 15;

    const v16bf aq0 = load_a16x32(Qh, DMODEL, m0, 0,  lane);
    const v16bf aq1 = load_a16x32(Qh, DMODEL, m0, 32, lane);

    v8f o0 = {}, o1 = {}, o2 = {}, o3 = {};
    float mrow[8], lrow[8];
#pragma unroll
    for (int r = 0; r < 8; ++r) { mrow[r] = -3.0e38f; lrow[r] = 0.0f; }

    for (int n0 = 0; n0 <= m0 + 15; n0 += 32) {        // causal key blocks of 32
        if (n0 + 32 <= m0 + 15) {                      // prefetch next key block rows
            __builtin_prefetch(Kh + (size_t)(n0 + 32 + lane) * DMODEL, 0, 0);
            __builtin_prefetch(Vh + (size_t)(n0 + 32 + lane) * DMODEL, 0, 0);
        }
        v8f s0 = {}, s1 = {};
        {
            v16bf bk = load_b_nk(Kh, DMODEL, n0, 0, lane);
            s0 = wmma_bf16(aq0, bk, s0);
            bk = load_b_nk(Kh, DMODEL, n0, 32, lane);
            s0 = wmma_bf16(aq1, bk, s0);
            bk = load_b_nk(Kh, DMODEL, n0 + 16, 0, lane);
            s1 = wmma_bf16(aq0, bk, s1);
            bk = load_b_nk(Kh, DMODEL, n0 + 16, 32, lane);
            s1 = wmma_bf16(aq1, bk, s1);
        }
        float alpha[8];
#pragma unroll
        for (int r = 0; r < 8; ++r) {
            const int qi = m0 + r + 8 * g;
            const float x0 = (n0 + nl      <= qi) ? s0[r] : -1.0e30f;
            const float x1 = (n0 + 16 + nl <= qi) ? s1[r] : -1.0e30f;
            float mx = fmaxf(x0, x1);
#pragma unroll
            for (int off = 1; off < 16; off <<= 1)
                mx = fmaxf(mx, __shfl_xor(mx, off, 32));
            const float mnew = fmaxf(mrow[r], mx);
            const float al = __expf(mrow[r] - mnew);
            const float p0 = __expf(x0 - mnew);
            const float p1 = __expf(x1 - mnew);
            float ps = p0 + p1;
#pragma unroll
            for (int off = 1; off < 16; off <<= 1)
                ps += __shfl_xor(ps, off, 32);
            mrow[r]  = mnew;
            lrow[r]  = lrow[r] * al + ps;
            alpha[r] = al;
            pbuf[(r + 8 * g) * 32 + nl]      = f32_to_bf16_rne(p0);
            pbuf[(r + 8 * g) * 32 + 16 + nl] = f32_to_bf16_rne(p1);
        }
#pragma unroll
        for (int r = 0; r < 8; ++r) {
            o0[r] *= alpha[r]; o1[r] *= alpha[r];
            o2[r] *= alpha[r]; o3[r] *= alpha[r];
        }
        __syncthreads();
        const v16bf ap = load_a16x32(pbuf, 32, 0, 0, lane);
        __syncthreads();
        o0 = wmma_bf16(ap, load_b_kn(Vh, DMODEL, n0, 0,  lane), o0);
        o1 = wmma_bf16(ap, load_b_kn(Vh, DMODEL, n0, 16, lane), o1);
        o2 = wmma_bf16(ap, load_b_kn(Vh, DMODEL, n0, 32, lane), o2);
        o3 = wmma_bf16(ap, load_b_kn(Vh, DMODEL, n0, 48, lane), o3);
    }
#pragma unroll
    for (int r = 0; r < 8; ++r) {
        const int qi = m0 + r + 8 * g;
        const float inv = 1.0f / lrow[r];
        unsigned short* orow = O + (size_t)(b * SEQ + qi) * DMODEL + h * DHEAD;
        orow[0 * 16 + nl] = f32_to_bf16_rne(o0[r] * inv);
        orow[1 * 16 + nl] = f32_to_bf16_rne(o1[r] * inv);
        orow[2 * 16 + nl] = f32_to_bf16_rne(o2[r] * inv);
        orow[3 * 16 + nl] = f32_to_bf16_rne(o3[r] * inv);
    }
}

// ---------- launch ----------

extern "C" void kernel_launch(void* const* d_in, const int* in_sizes, int n_in,
                              void* d_out, int out_size, void* d_ws, size_t ws_size,
                              hipStream_t stream) {
    (void)in_sizes; (void)n_in; (void)out_size; (void)ws_size;
    const float* x  = (const float*)d_in[0];
    // d_in[1] = attn_mask (causal; implemented analytically)
    const float* Wq = (const float*)d_in[2];
    const float* Wk = (const float*)d_in[3];
    const float* Wv = (const float*)d_in[4];
    const float* Wo = (const float*)d_in[5];
    float* out = (float*)d_out;

    char* ws = (char*)d_ws;
    const size_t MB = 1024 * 1024;
    unsigned short* xb  = (unsigned short*)(ws + 0 * MB);   // 8 MiB (4096x1024 bf16)
    unsigned short* wqb = (unsigned short*)(ws + 8 * MB);   // 2 MiB each
    unsigned short* wkb = (unsigned short*)(ws + 10 * MB);
    unsigned short* wvb = (unsigned short*)(ws + 12 * MB);
    unsigned short* wob = (unsigned short*)(ws + 14 * MB);
    unsigned short* qb  = (unsigned short*)(ws + 16 * MB);  // 8 MiB each
    unsigned short* kb  = (unsigned short*)(ws + 24 * MB);
    unsigned short* vb  = (unsigned short*)(ws + 32 * MB);
    unsigned short* ob  = (unsigned short*)(ws + 40 * MB);  // ends at 48 MiB

    const int M = BATCH * SEQ;   // 4096
    const int D = DMODEL;        // 1024

    cvt_bf16_kernel<<<dim3((M * D + 255) / 256), 256, 0, stream>>>(x,  xb,  M * D);
    cvt_bf16_kernel<<<dim3((D * D + 255) / 256), 256, 0, stream>>>(Wq, wqb, D * D);
    cvt_bf16_kernel<<<dim3((D * D + 255) / 256), 256, 0, stream>>>(Wk, wkb, D * D);
    cvt_bf16_kernel<<<dim3((D * D + 255) / 256), 256, 0, stream>>>(Wv, wvb, D * D);
    cvt_bf16_kernel<<<dim3((D * D + 255) / 256), 256, 0, stream>>>(Wo, wob, D * D);

    const dim3 ggrid(D / 64, M / 128);     // 16 x 32 blocks of 128 threads
    gemm_bt_kernel<<<ggrid, 128, 0, stream>>>(xb, wqb, qb, nullptr, M, D, D, 0.125f);
    gemm_bt_kernel<<<ggrid, 128, 0, stream>>>(xb, wkb, kb, nullptr, M, D, D, 1.0f);
    gemm_bt_kernel<<<ggrid, 128, 0, stream>>>(xb, wvb, vb, nullptr, M, D, D, 1.0f);

    attn_kernel<<<dim3(SEQ / 16, BATCH * NHEAD), 32, 0, stream>>>(qb, kb, vb, ob);

    gemm_bt_kernel<<<ggrid, 128, 0, stream>>>(ob, wob, nullptr, out, M, D, D, 1.0f);
}
